// AceStepAttention_82798379532342
// MI455X (gfx1250) — matrix operand order repaired and checked by reading, not compiled, not run on hardware
//
#include <hip/hip_runtime.h>
#include <hip/hip_bf16.h>
#include <math.h>

// ---------------------------------------------------------------------------
// AceStep attention block for MI455X (gfx1250, wave32, WMMA bf16)
// B=2, S=2048, HS=2048, H=16, KV=4, D=128
// ---------------------------------------------------------------------------

typedef __bf16 bf16;
typedef __attribute__((ext_vector_type(16))) __bf16 v16bf;
typedef __attribute__((ext_vector_type(8)))  float  v8f;

#define B_   2
#define S_   2048
#define HS_  2048
#define H_   16
#define KV_  4
#define D_   128
#define QSTR (H_ * D_)   // 2048
#define KSTR (KV_ * D_)  // 512

static __device__ __forceinline__ v8f wmma_bf16(v16bf a, v16bf b, v8f c) {
  return __builtin_amdgcn_wmma_f32_16x16x32_bf16(false, a, false, b,
                                                 (short)0, c, false, false);
}

// ---------------------------------------------------------------------------
// async global->LDS DMA (gfx1250 ASYNCcnt path), with safe fallback
// ---------------------------------------------------------------------------
#if __has_builtin(__builtin_amdgcn_global_load_async_to_lds_b128) && \
    __has_builtin(__builtin_amdgcn_s_wait_asynccnt)
#define ASYNC_LDS 1
#else
#define ASYNC_LDS 0
#endif

typedef int v4i __attribute__((vector_size(16)));
typedef __attribute__((address_space(1))) v4i* as1_v4i_p;
typedef __attribute__((address_space(3))) v4i* as3_v4i_p;

static __device__ __forceinline__ void copy16_g2l(const void* g, void* l) {
#if ASYNC_LDS
  __builtin_amdgcn_global_load_async_to_lds_b128((as1_v4i_p)g, (as3_v4i_p)l,
                                                 0, 0);
#else
  *reinterpret_cast<uint4*>(l) = *reinterpret_cast<const uint4*>(g);
#endif
}

#if ASYNC_LDS
#define WAIT_ASYNC(n) __builtin_amdgcn_s_wait_asynccnt(n)
#else
#define WAIT_ASYNC(n) ((void)0)
#endif

// ---------------------------------------------------------------------------
// 1) f32 -> bf16 conversion (grid-strided)
// ---------------------------------------------------------------------------
__global__ void cvt_f32_bf16(const float* __restrict__ in, bf16* __restrict__ out,
                             long n) {
  long i = (long)blockIdx.x * blockDim.x + threadIdx.x;
  long stride = (long)gridDim.x * blockDim.x;
  for (; i < n; i += stride) out[i] = (bf16)in[i];
}

// ---------------------------------------------------------------------------
// 2) C[M,N](f32) = A[M,K](bf16) @ W[N,K](bf16)^T
//    block: 256 threads = 8 waves; C tile 128x128; K-step 32
//    double-buffered LDS, async DMA staging overlaps the 8 WMMAs/wave/step
// ---------------------------------------------------------------------------
#define TM 128
#define TN 128
#define TK 32

__global__ __launch_bounds__(256) void gemm_bf16_nt(
    const bf16* __restrict__ A, const bf16* __restrict__ W,
    float* __restrict__ C, int M, int N, int K) {
  __shared__ bf16 As[2][TM][TK];   // 16 KB
  __shared__ bf16 Ws[2][TN][TK];   // 16 KB

  const int tid  = threadIdx.x;
  const int lane = tid & 31;
  const int wave = tid >> 5;            // 0..7
  const int wm   = (wave >> 2) * 64;    // 0,64
  const int wn   = (wave & 3) * 32;     // 0,32,64,96
  const int bm   = blockIdx.y * TM;
  const int bn   = blockIdx.x * TN;

  const int lrow = tid >> 1;            // 0..127
  const int lcol = (tid & 1) * 16;      // 0,16

  v8f acc[4][2];
  v8f zero = {};
#pragma unroll
  for (int i = 0; i < 4; ++i)
#pragma unroll
    for (int j = 0; j < 2; ++j) acc[i][j] = zero;

  const int mlo   = lane & 15;
  const int ahalf = (lane >> 4) * 8;    // unused; A frags also use bhalf
  const int bhalf = (lane >> 4) * 16;   // B frag K offset

  auto stage = [&](int p, int k0) {
    copy16_g2l(&A[(size_t)(bm + lrow) * K + k0 + lcol], &As[p][lrow][lcol]);
    copy16_g2l(&W[(size_t)(bn + lrow) * K + k0 + lcol], &Ws[p][lrow][lcol]);
  };

  stage(0, 0);
  int p = 0;
  for (int k0 = 0; k0 < K; k0 += TK, p ^= 1) {
    const bool more = (k0 + TK) < K;
    if (more) {
      stage(p ^ 1, k0 + TK);   // 2 async ops for the NEXT tile stay in flight
      WAIT_ASYNC(2);           // current tile's DMA complete (in-order)
    } else {
      WAIT_ASYNC(0);
    }
    __syncthreads();

    // A fragments (16x32), one per 16-row subtile
    v16bf af[4];
#pragma unroll
    for (int sm = 0; sm < 4; ++sm) {
      const int m = wm + sm * 16 + mlo;
#pragma unroll
      for (int e = 0; e < 8; ++e) af[sm][e]     = As[p][m][bhalf + e];
#pragma unroll
      for (int e = 0; e < 8; ++e) af[sm][8 + e] = As[p][m][bhalf + 8 + e];
    }
    // B fragments (32x16), one per 16-col subtile
    v16bf bfgs[2];
#pragma unroll
    for (int sn = 0; sn < 2; ++sn) {
      const int n = wn + sn * 16 + mlo;
#pragma unroll
      for (int e = 0; e < 16; ++e) bfgs[sn][e] = Ws[p][n][bhalf + e];
    }
#pragma unroll
    for (int sm = 0; sm < 4; ++sm)
#pragma unroll
      for (int sn = 0; sn < 2; ++sn)
        acc[sm][sn] = wmma_bf16(af[sm], bfgs[sn], acc[sm][sn]);

    __syncthreads();
  }

  // store: C/D layout -> VGPR r: rows r / r+8 by lane half, col = lane&15
  const int rbase = (lane >= 16) ? 8 : 0;
#pragma unroll
  for (int sm = 0; sm < 4; ++sm)
#pragma unroll
    for (int sn = 0; sn < 2; ++sn) {
      const int cn = bn + wn + sn * 16 + mlo;
#pragma unroll
      for (int r = 0; r < 8; ++r) {
        const int cm = bm + wm + sm * 16 + r + rbase;
        C[(size_t)cm * N + cn] = acc[sm][sn][r];
      }
    }
}

// ---------------------------------------------------------------------------
// 3) fused RMSNorm + RoPE + bf16 convert, one (b,s,head) row of D=128 / block
// ---------------------------------------------------------------------------
__global__ __launch_bounds__(128) void rmsnorm_rope_cvt(
    const float* __restrict__ x, const float* __restrict__ cs,
    const float* __restrict__ sn, const float* __restrict__ w,
    bf16* __restrict__ out, int NH) {
  __shared__ float red[128];
  __shared__ float xn[128];
  const int d   = threadIdx.x;
  const int row = blockIdx.x;          // b*S*NH flat
  const int hh  = row % NH;
  const int bs  = row / NH;
  const int s   = bs % S_;

  const size_t base = (size_t)bs * NH * D_ + (size_t)hh * D_;
  const float v = x[base + d];

  red[d] = v * v;
  __syncthreads();
#pragma unroll
  for (int off = 64; off > 0; off >>= 1) {
    if (d < off) red[d] += red[d + off];
    __syncthreads();
  }
  const float rinv = rsqrtf(red[0] / (float)D_ + 1e-6f);
  __syncthreads();

  const float y = v * rinv * w[d];
  xn[d] = y;
  __syncthreads();
  const float rot = (d < 64) ? -xn[d + 64] : xn[d - 64];
  const float o = y * cs[(size_t)s * D_ + d] + rot * sn[(size_t)s * D_ + d];
  out[base + d] = (bf16)o;
}

// ---------------------------------------------------------------------------
// 4) flash attention: 4 waves/block, each wave one 16-row q tile; the block
//    shares double-buffered 32x128 K/V chunks staged in LDS via async DMA.
//    Q/K/V bf16; O bf16 [B*S, H*D]
// ---------------------------------------------------------------------------
__global__ __launch_bounds__(128) void flash_attn(
    const bf16* __restrict__ Q, const bf16* __restrict__ Kb,
    const bf16* __restrict__ V, bf16* __restrict__ O) {
  const int tid  = threadIdx.x;
  const int lane = tid & 31;
  const int wv   = tid >> 5;       // 0..3
  const int qt   = blockIdx.x;     // 0..S/64-1
  const int h    = blockIdx.y;     // 0..H-1
  const int b    = blockIdx.z;     // 0..B-1
  const int kvh  = h >> 2;         // H/KV = 4

  const int mlo   = lane & 15;
  const int ahalf = (lane >> 4) * 8;
  const int bhalf = (lane >> 4) * 16;
  const int rbase = (lane >= 16) ? 8 : 0;

  const size_t qrow0 = (size_t)b * S_ + (size_t)qt * 64 + (size_t)wv * 16;

  __shared__ bf16 Ks[2][32][D_];       // 16 KB
  __shared__ bf16 Vs[2][32][D_];       // 16 KB
  __shared__ bf16 Pl[4][16][32];       // 4 KB

  // staging: 128 threads x 64B per matrix per chunk (coalesced 16B units)
  const int srow = tid >> 2;           // 0..31
  const int scol = (tid & 3) * 32;     // 0,32,64,96
  const size_t kvbase = (size_t)b * S_ * KSTR + (size_t)kvh * D_;

  auto stage = [&](int p, int kv0) {
    const bf16* kg = Kb + kvbase + (size_t)(kv0 + srow) * KSTR + scol;
    const bf16* vg = V  + kvbase + (size_t)(kv0 + srow) * KSTR + scol;
#pragma unroll
    for (int j = 0; j < 4; ++j) {
      copy16_g2l(kg + j * 8, &Ks[p][srow][scol + j * 8]);
      copy16_g2l(vg + j * 8, &Vs[p][srow][scol + j * 8]);
    }
  };

  // Q fragments: 4 x (16x32) covering D=128
  v16bf qf[4];
#pragma unroll
  for (int f = 0; f < 4; ++f) {
    const bf16* qp = Q + (qrow0 + mlo) * QSTR + (size_t)h * D_ + f * 32;
#pragma unroll
    for (int e = 0; e < 8; ++e) qf[f][e]     = qp[bhalf + e];
#pragma unroll
    for (int e = 0; e < 8; ++e) qf[f][8 + e] = qp[bhalf + 8 + e];
  }

  v8f zero = {};
  v8f oacc[8];
#pragma unroll
  for (int i = 0; i < 8; ++i) oacc[i] = zero;
  float mrow[8], lrow[8];
#pragma unroll
  for (int r = 0; r < 8; ++r) { mrow[r] = -INFINITY; lrow[r] = 0.f; }

  const float scale = 0.0883883476483184f;  // 1/sqrt(128)
  const float log2e = 1.4426950408889634f;

  stage(0, 0);
  int p = 0;
  for (int kv0 = 0; kv0 < S_; kv0 += 32, p ^= 1) {
    const bool more = (kv0 + 32) < S_;
    if (more) {
      stage(p ^ 1, kv0 + 32);  // 8 async ops for NEXT chunk stay in flight
      WAIT_ASYNC(8);
    } else {
      WAIT_ASYNC(0);
    }
    __syncthreads();

    // ---- scores: two 16x16 tiles (kv cols kv0..kv0+31) from LDS ----
    v8f sc[2];
#pragma unroll
    for (int t = 0; t < 2; ++t) {
      v8f c = zero;
#pragma unroll
      for (int f = 0; f < 4; ++f) {
        v16bf bb;
        const int d0 = f * 32 + bhalf;
#pragma unroll
        for (int e = 0; e < 16; ++e) bb[e] = Ks[p][t * 16 + mlo][d0 + e];
        c = wmma_bf16(qf[f], bb, c);
      }
#pragma unroll
      for (int r = 0; r < 8; ++r) c[r] *= scale;
      sc[t] = c;
    }

    // ---- online softmax (row reductions inside 16-lane halves) ----
    float mnew[8], corr[8];
#pragma unroll
    for (int r = 0; r < 8; ++r) {
      float v = fmaxf(sc[0][r], sc[1][r]);
#pragma unroll
      for (int off = 1; off < 16; off <<= 1) v = fmaxf(v, __shfl_xor(v, off));
      mnew[r] = fmaxf(mrow[r], v);
      corr[r] = exp2f((mrow[r] - mnew[r]) * log2e);
      mrow[r] = mnew[r];
      lrow[r] *= corr[r];
    }
#pragma unroll
    for (int i = 0; i < 8; ++i)
#pragma unroll
      for (int r = 0; r < 8; ++r) oacc[i][r] *= corr[r];

    // ---- p = exp(s - m); row sums; relayout via LDS ----
#pragma unroll
    for (int t = 0; t < 2; ++t)
#pragma unroll
      for (int r = 0; r < 8; ++r) {
        const float pv = exp2f((sc[t][r] - mrow[r]) * log2e);
        float ps = pv;
#pragma unroll
        for (int off = 1; off < 16; off <<= 1) ps += __shfl_xor(ps, off);
        lrow[r] += ps;  // full row-sum replicated in every lane of the half
        Pl[wv][r + rbase][t * 16 + mlo] = (bf16)pv;
      }
    __syncthreads();

    // P as A fragment (16x32)
    v16bf pa;
#pragma unroll
    for (int e = 0; e < 8; ++e) pa[e]     = Pl[wv][mlo][bhalf + e];
#pragma unroll
    for (int e = 0; e < 8; ++e) pa[8 + e] = Pl[wv][mlo][bhalf + 8 + e];

    // ---- O += P @ V : 8 tiles across D, V from LDS ----
#pragma unroll
    for (int dt = 0; dt < 8; ++dt) {
      v16bf vb;
      const int n = dt * 16 + mlo;
#pragma unroll
      for (int e = 0; e < 16; ++e) vb[e] = Vs[p][bhalf + e][n];
      oacc[dt] = wmma_bf16(pa, vb, oacc[dt]);
    }
    __syncthreads();
  }

  // ---- finalize ----
#pragma unroll
  for (int dt = 0; dt < 8; ++dt)
#pragma unroll
    for (int r = 0; r < 8; ++r) {
      const float o = oacc[dt][r] / lrow[r];
      const size_t row = qrow0 + r + rbase;
      O[row * QSTR + (size_t)h * D_ + dt * 16 + mlo] = (bf16)o;
    }
}

// ---------------------------------------------------------------------------
// launch
// ---------------------------------------------------------------------------
static inline size_t align256(size_t x) { return (x + 255) & ~(size_t)255; }

extern "C" void kernel_launch(void* const* d_in, const int* in_sizes, int n_in,
                              void* d_out, int out_size, void* d_ws, size_t ws_size,
                              hipStream_t stream) {
  (void)in_sizes; (void)n_in; (void)out_size; (void)ws_size;

  const float* hidden = (const float*)d_in[0];
  const float* cosb   = (const float*)d_in[1];
  const float* sinb   = (const float*)d_in[2];
  const float* Wq     = (const float*)d_in[3];
  const float* Wk     = (const float*)d_in[4];
  const float* Wv     = (const float*)d_in[5];
  const float* Wo     = (const float*)d_in[6];
  const float* nqw    = (const float*)d_in[7];
  const float* nkw    = (const float*)d_in[8];
  float* out = (float*)d_out;

  const long BS = (long)B_ * S_;          // 4096
  char* ws = (char*)d_ws;
  size_t off = 0;
  auto carve = [&](size_t bytes) { char* p = ws + off; off += align256(bytes); return p; };

  bf16* h_bf   = (bf16*)carve(BS * HS_ * 2);
  bf16* Wq_bf  = (bf16*)carve((size_t)QSTR * HS_ * 2);
  bf16* Wk_bf  = (bf16*)carve((size_t)KSTR * HS_ * 2);
  bf16* Wv_bf  = (bf16*)carve((size_t)KSTR * HS_ * 2);
  bf16* Wo_bf  = (bf16*)carve((size_t)HS_ * QSTR * 2);
  float* q_f   = (float*)carve(BS * QSTR * 4);
  float* k_f   = (float*)carve(BS * KSTR * 4);
  float* v_f   = (float*)carve(BS * KSTR * 4);
  bf16* q_bf   = (bf16*)carve(BS * QSTR * 2);
  bf16* k_bf   = (bf16*)carve(BS * KSTR * 2);
  bf16* v_bf   = (bf16*)carve(BS * KSTR * 2);
  bf16* att_bf = (bf16*)carve(BS * QSTR * 2);

  // 1) convert inputs to bf16
  hipLaunchKernelGGL(cvt_f32_bf16, dim3(2048), dim3(256), 0, stream,
                     hidden, h_bf, BS * (long)HS_);
  hipLaunchKernelGGL(cvt_f32_bf16, dim3(2048), dim3(256), 0, stream,
                     Wq, Wq_bf, (long)QSTR * HS_);
  hipLaunchKernelGGL(cvt_f32_bf16, dim3(1024), dim3(256), 0, stream,
                     Wk, Wk_bf, (long)KSTR * HS_);
  hipLaunchKernelGGL(cvt_f32_bf16, dim3(1024), dim3(256), 0, stream,
                     Wv, Wv_bf, (long)KSTR * HS_);
  hipLaunchKernelGGL(cvt_f32_bf16, dim3(2048), dim3(256), 0, stream,
                     Wo, Wo_bf, (long)HS_ * QSTR);

  // 2) projections: q/k/v = hidden @ W^T
  hipLaunchKernelGGL(gemm_bf16_nt, dim3(QSTR / TN, BS / TM), dim3(256), 0, stream,
                     h_bf, Wq_bf, q_f, (int)BS, QSTR, HS_);
  hipLaunchKernelGGL(gemm_bf16_nt, dim3(KSTR / TN, BS / TM), dim3(256), 0, stream,
                     h_bf, Wk_bf, k_f, (int)BS, KSTR, HS_);
  hipLaunchKernelGGL(gemm_bf16_nt, dim3(KSTR / TN, BS / TM), dim3(256), 0, stream,
                     h_bf, Wv_bf, v_f, (int)BS, KSTR, HS_);

  // 3) RMSNorm + RoPE (q, k); plain convert (v)
  hipLaunchKernelGGL(rmsnorm_rope_cvt, dim3((unsigned)(BS * H_)), dim3(128), 0, stream,
                     q_f, cosb, sinb, nqw, q_bf, H_);
  hipLaunchKernelGGL(rmsnorm_rope_cvt, dim3((unsigned)(BS * KV_)), dim3(128), 0, stream,
                     k_f, cosb, sinb, nkw, k_bf, KV_);
  hipLaunchKernelGGL(cvt_f32_bf16, dim3(1024), dim3(256), 0, stream,
                     v_f, v_bf, BS * (long)KSTR);

  // 4) flash attention (64 q rows per block, shared K/V chunks in LDS)
  hipLaunchKernelGGL(flash_attn, dim3(S_ / 64, H_, B_), dim3(128), 0, stream,
                     q_bf, k_bf, v_bf, att_bf);

  // 5) output projection: out = attn @ Wo^T  (f32 result)
  hipLaunchKernelGGL(gemm_bf16_nt, dim3(HS_ / TN, BS / TM), dim3(256), 0, stream,
                     att_bf, Wo_bf, out, (int)BS, HS_, QSTR);
}
